// CognitiveWorkspaceTransformer_34299608825948
// MI455X (gfx1250) — matrix operand order repaired and checked by reading, not compiled, not run on hardware
//
#include <hip/hip_runtime.h>

// ---------------------------------------------------------------------------
// MI455X (gfx1250) implementation.
// out = (S + concat(w_spoke,w_hub_priv,w_hub_shared,tag)) @ W_read^T
// k   = latent @ Wk^T ;  v = latent @ Wv^T     (cache := latent, not returned)
//
// Strategy: memory-bound (~710MB traffic -> ~30us @ 23.3TB/s), so use the
// fast bf16 WMMA path with split-precision (hi+lo) to retain ~fp32 accuracy:
//   x = hi(x) + lo(x);  A*B ~= Ah*Bh + Ah*Bl + Al*Bh  (f32 accumulate)
// Weights are pre-split into bf16 hi/lo in B-fragment register order in d_ws
// so the GEMM inner loop is pure global_load_b128 / ds_load_b128 / v_wmma.
// ---------------------------------------------------------------------------

typedef __attribute__((ext_vector_type(16))) __bf16 v16bf;
typedef __attribute__((ext_vector_type(8)))  __bf16 v8bf;
typedef __attribute__((ext_vector_type(4)))  __bf16 v4bf;
typedef __attribute__((ext_vector_type(8)))  float  v8f;
typedef __attribute__((ext_vector_type(4)))  float  v4f;

#define D_STATE  1088
#define D_MODEL  1024
#define D_LATENT 128
#define ROWS     32768          // B*T = 8*4096
#define BM       128
#define BN       128
#define LDA      40             // padded bf16 stride for 32-K LDS tile (80B, 16B-aligned)
#define LDK      136            // padded bf16 stride for 128-K LDS tile (272B, 16B-aligned)

__device__ __forceinline__ v16bf join16(v8bf a, v8bf b) {
  union { v16bf v; v8bf h[2]; } u;
  u.h[0] = a; u.h[1] = b;
  return u.v;
}

__device__ __forceinline__ v8f wmma_bf16(v16bf a, v16bf b, v8f c) {
  // (neg_a, A, neg_b, B, c_mod, C, reuse_a, reuse_b)
  return __builtin_amdgcn_wmma_f32_16x16x32_bf16(false, a, false, b, (short)0, c,
                                                 false, false);
}

// ---------------------------------------------------------------------------
// Pre-split weight W (n-major: W[n*Ktot + k], n in [0,1024)) into bf16 hi/lo
// laid out in B-fragment register order for v_wmma_f32_16x16x32_bf16:
//   id = ((kc*64 + nt)*32 + lane)*16 + e
//   lane n%16 (+16 for khalf=1) holds column n, K = kc*32 + (e<8?0:16) + khalf*8 + (e&7).
// ---------------------------------------------------------------------------
__global__ void prep_frag(const float* __restrict__ W, int Ktot,
                          __bf16* __restrict__ hi, __bf16* __restrict__ lo,
                          int total) {
  int id = blockIdx.x * blockDim.x + threadIdx.x;
  if (id >= total) return;
  int e    = id & 15;
  int lane = (id >> 4) & 31;
  int nt   = (id >> 9) & 63;
  int kc   = id >> 15;
  int khalf = lane >> 4;
  int K = kc * 32 + (e < 8 ? 0 : 16) + khalf * 8 + (e & 7);
  int n = nt * 16 + (lane & 15);
  float v = W[(size_t)n * Ktot + K];
  __bf16 h = (__bf16)v;
  float r = v - (float)h;
  hi[id] = h;
  lo[id] = (__bf16)r;
}

// ---------------------------------------------------------------------------
// out[m,n] = sum_k (S[m,k] + w_seg(k)[m]) * W[n,k]   (32768x1024, K=1088)
// 128x128 tile / workgroup, 8 waves (4 x 2 of 32x64), K-chunks of 32.
// Segment boundaries (512,768,1024) are 32-aligned -> one w pointer per chunk.
// ---------------------------------------------------------------------------
__global__ void __launch_bounds__(256)
ws_read_gemm(const float* __restrict__ S,
             const float* __restrict__ w_spoke,
             const float* __restrict__ w_hub_priv,
             const float* __restrict__ w_hub_shared,
             const float* __restrict__ tag,
             const __bf16* __restrict__ Whi,
             const __bf16* __restrict__ Wlo,
             float* __restrict__ out) {
  __shared__ __attribute__((aligned(16))) __bf16 Ah[BM * LDA];
  __shared__ __attribute__((aligned(16))) __bf16 Al[BM * LDA];

  const int tid  = threadIdx.x;
  const int lane = tid & 31;
  const int wave = tid >> 5;
  const int wm   = wave >> 1;            // 0..3 -> 32 rows each
  const int wn   = wave & 1;             // 0..1 -> 64 cols each
  const int row0 = blockIdx.x * BM;
  const int col0 = blockIdx.y * BN;
  const int nt0  = col0 >> 4;

  const int lm = tid >> 1;               // loader row 0..127
  const int kq = (tid & 1) * 16;         // loader k-offset 0/16

  v8f acc[2][4];
#pragma unroll
  for (int a = 0; a < 2; ++a)
#pragma unroll
    for (int b = 0; b < 4; ++b)
#pragma unroll
      for (int i = 0; i < 8; ++i) acc[a][b][i] = 0.0f;

  for (int kc = 0; kc < 34; ++kc) {
    const int kbase = kc * 32;
    const float* wp; int wstride, woff;
    if (kbase < 512)       { wp = w_spoke;      wstride = 512; woff = kbase; }
    else if (kbase < 768)  { wp = w_hub_priv;   wstride = 256; woff = kbase - 512; }
    else if (kbase < 1024) { wp = w_hub_shared; wstride = 256; woff = kbase - 768; }
    else                   { wp = tag;          wstride = 64;  woff = kbase - 1024; }

    const float* sp = S + (size_t)(row0 + lm) * D_STATE + kbase + kq;
    const float* wq = wp + (size_t)(row0 + lm) * wstride + woff + kq;
    __bf16* ah = &Ah[lm * LDA + kq];
    __bf16* al = &Al[lm * LDA + kq];
    if (kc + 1 < 34) __builtin_prefetch(sp + 32, 0, 1);   // next K-chunk of S
#pragma unroll
    for (int i = 0; i < 4; ++i) {
      v4f s4 = *(const v4f*)(sp + 4 * i);
      v4f w4 = *(const v4f*)(wq + 4 * i);
      v4bf h4, l4;
#pragma unroll
      for (int j = 0; j < 4; ++j) {
        float x = s4[j] + w4[j];
        __bf16 h = (__bf16)x;
        h4[j] = h;
        l4[j] = (__bf16)(x - (float)h);
      }
      *(v4bf*)(ah + 4 * i) = h4;
      *(v4bf*)(al + 4 * i) = l4;
    }
    __syncthreads();

    // A fragments (16-bit A layout: K = (e<8?0:16) + khalf*8 + (e&7))
    const int khalf = lane >> 4;
    const int am    = lane & 15;
    v16bf afh[2], afl[2];
#pragma unroll
    for (int sm = 0; sm < 2; ++sm) {
      int r = wm * 32 + sm * 16 + am;
      const __bf16* bh = &Ah[r * LDA + khalf * 8];
      afh[sm] = join16(*(const v8bf*)bh, *(const v8bf*)(bh + 16));
      const __bf16* bl = &Al[r * LDA + khalf * 8];
      afl[sm] = join16(*(const v8bf*)bl, *(const v8bf*)(bl + 16));
    }

#pragma unroll
    for (int sn = 0; sn < 4; ++sn) {
      int nt = nt0 + wn * 4 + sn;
      size_t fb = ((size_t)(kc * 64 + nt) << 9) + ((size_t)lane << 4);
      v16bf bh = join16(*(const v8bf*)(Whi + fb), *(const v8bf*)(Whi + fb + 8));
      v16bf bl = join16(*(const v8bf*)(Wlo + fb), *(const v8bf*)(Wlo + fb + 8));
#pragma unroll
      for (int sm = 0; sm < 2; ++sm) {
        acc[sm][sn] = wmma_bf16(afh[sm], bh, acc[sm][sn]);
        acc[sm][sn] = wmma_bf16(afl[sm], bh, acc[sm][sn]);
        acc[sm][sn] = wmma_bf16(afh[sm], bl, acc[sm][sn]);
      }
    }
    __syncthreads();
  }

  // C layout: element i -> row = i + (lane>=16 ? 8 : 0), col = lane&15
#pragma unroll
  for (int sm = 0; sm < 2; ++sm) {
    int mbase = row0 + wm * 32 + sm * 16 + (lane >> 4) * 8;
#pragma unroll
    for (int sn = 0; sn < 4; ++sn) {
      int n = col0 + wn * 64 + sn * 16 + (lane & 15);
#pragma unroll
      for (int i = 0; i < 8; ++i)
        out[(size_t)(mbase + i) * D_MODEL + n] = acc[sm][sn][i];
    }
  }
}

// ---------------------------------------------------------------------------
// k = latent @ Wk^T and v = latent @ Wv^T (K=128). Stage the full 128x128
// latent tile (hi/lo) in LDS once, reuse it for both output GEMMs.
// ---------------------------------------------------------------------------
__global__ void __launch_bounds__(256)
kv_gemm(const float* __restrict__ latent,
        const __bf16* __restrict__ Khi, const __bf16* __restrict__ Klo,
        const __bf16* __restrict__ Vhi, const __bf16* __restrict__ Vlo,
        float* __restrict__ outk, float* __restrict__ outv) {
  __shared__ __attribute__((aligned(16))) __bf16 Ah[BM * LDK];
  __shared__ __attribute__((aligned(16))) __bf16 Al[BM * LDK];

  const int tid  = threadIdx.x;
  const int lane = tid & 31;
  const int wave = tid >> 5;
  const int wm   = wave >> 1;
  const int wn   = wave & 1;
  const int row0 = blockIdx.x * BM;
  const int col0 = blockIdx.y * BN;
  const int nt0  = col0 >> 4;

  const int lm = tid >> 1;
  const int kq = (tid & 1) * 64;
  {
    const float* sp = latent + (size_t)(row0 + lm) * D_LATENT + kq;
    __bf16* ah = &Ah[lm * LDK + kq];
    __bf16* al = &Al[lm * LDK + kq];
#pragma unroll
    for (int i = 0; i < 16; ++i) {
      v4f s4 = *(const v4f*)(sp + 4 * i);
      v4bf h4, l4;
#pragma unroll
      for (int j = 0; j < 4; ++j) {
        float x = s4[j];
        __bf16 h = (__bf16)x;
        h4[j] = h;
        l4[j] = (__bf16)(x - (float)h);
      }
      *(v4bf*)(ah + 4 * i) = h4;
      *(v4bf*)(al + 4 * i) = l4;
    }
  }
  __syncthreads();

  const int khalf = lane >> 4;
  const int am    = lane & 15;

#pragma unroll
  for (int which = 0; which < 2; ++which) {
    const __bf16* Bh = which ? Vhi : Khi;
    const __bf16* Bl = which ? Vlo : Klo;
    float* o = which ? outv : outk;

    v8f acc[2][4];
#pragma unroll
    for (int a = 0; a < 2; ++a)
#pragma unroll
      for (int b = 0; b < 4; ++b)
#pragma unroll
        for (int i = 0; i < 8; ++i) acc[a][b][i] = 0.0f;

#pragma unroll
    for (int kc = 0; kc < 4; ++kc) {
      v16bf afh[2], afl[2];
#pragma unroll
      for (int sm = 0; sm < 2; ++sm) {
        int r = wm * 32 + sm * 16 + am;
        const __bf16* bh = &Ah[r * LDK + kc * 32 + khalf * 8];
        afh[sm] = join16(*(const v8bf*)bh, *(const v8bf*)(bh + 16));
        const __bf16* bl = &Al[r * LDK + kc * 32 + khalf * 8];
        afl[sm] = join16(*(const v8bf*)bl, *(const v8bf*)(bl + 16));
      }
#pragma unroll
      for (int sn = 0; sn < 4; ++sn) {
        int nt = nt0 + wn * 4 + sn;
        size_t fb = ((size_t)(kc * 64 + nt) << 9) + ((size_t)lane << 4);
        v16bf bh = join16(*(const v8bf*)(Bh + fb), *(const v8bf*)(Bh + fb + 8));
        v16bf bl = join16(*(const v8bf*)(Bl + fb), *(const v8bf*)(Bl + fb + 8));
#pragma unroll
        for (int sm = 0; sm < 2; ++sm) {
          acc[sm][sn] = wmma_bf16(afh[sm], bh, acc[sm][sn]);
          acc[sm][sn] = wmma_bf16(afl[sm], bh, acc[sm][sn]);
          acc[sm][sn] = wmma_bf16(afh[sm], bl, acc[sm][sn]);
        }
      }
    }

#pragma unroll
    for (int sm = 0; sm < 2; ++sm) {
      int mbase = row0 + wm * 32 + sm * 16 + (lane >> 4) * 8;
#pragma unroll
      for (int sn = 0; sn < 4; ++sn) {
        int n = col0 + wn * 64 + sn * 16 + (lane & 15);
#pragma unroll
        for (int i = 0; i < 8; ++i)
          o[(size_t)(mbase + i) * D_MODEL + n] = acc[sm][sn][i];
      }
    }
  }
}

extern "C" void kernel_launch(void* const* d_in, const int* in_sizes, int n_in,
                              void* d_out, int out_size, void* d_ws, size_t ws_size,
                              hipStream_t stream) {
  const float* S            = (const float*)d_in[0];
  const float* w_spoke      = (const float*)d_in[1];
  const float* w_hub_priv   = (const float*)d_in[2];
  const float* w_hub_shared = (const float*)d_in[3];
  const float* tag          = (const float*)d_in[4];
  const float* W_read       = (const float*)d_in[5];
  // d_in[6] = cache (overwritten by latent in reference; not an output)
  const float* latent       = (const float*)d_in[7];
  const float* Wk           = (const float*)d_in[8];
  const float* Wv           = (const float*)d_in[9];

  float* out  = (float*)d_out;
  float* outk = out  + (size_t)ROWS * D_MODEL;
  float* outv = outk + (size_t)ROWS * D_MODEL;

  // Workspace layout (bf16 hi/lo weight fragments), ~5.5 MB total.
  __bf16* Whi = (__bf16*)d_ws;
  __bf16* Wlo = Whi + (size_t)D_STATE * D_MODEL;
  __bf16* Khi = Wlo + (size_t)D_STATE * D_MODEL;
  __bf16* Klo = Khi + (size_t)D_LATENT * D_MODEL;
  __bf16* Vhi = Klo + (size_t)D_LATENT * D_MODEL;
  __bf16* Vlo = Vhi + (size_t)D_LATENT * D_MODEL;

  const int totW  = D_STATE * D_MODEL;    // 1,114,112
  const int totKV = D_LATENT * D_MODEL;   // 131,072
  prep_frag<<<totW / 256, 256, 0, stream>>>(W_read, D_STATE, Whi, Wlo, totW);
  prep_frag<<<totKV / 256, 256, 0, stream>>>(Wk, D_LATENT, Khi, Klo, totKV);
  prep_frag<<<totKV / 256, 256, 0, stream>>>(Wv, D_LATENT, Vhi, Vlo, totKV);

  dim3 grid(ROWS / BM, D_MODEL / BN);     // 256 x 8
  ws_read_gemm<<<grid, 256, 0, stream>>>(S, w_spoke, w_hub_priv, w_hub_shared,
                                         tag, Whi, Wlo, out);
  kv_gemm<<<grid, 256, 0, stream>>>(latent, Khi, Klo, Vhi, Vlo, outk, outv);
}
